// CapsuleNetwork_57011395887203
// MI455X (gfx1250) — matrix operand order repaired and checked by reading, not compiled, not run on hardware
//
#include <hip/hip_runtime.h>
#include <hip/hip_bf16.h>
#include <math.h>

typedef __attribute__((ext_vector_type(16))) _Float16 v16h;
typedef __attribute__((ext_vector_type(8)))  _Float16 v8h;
typedef __attribute__((ext_vector_type(8)))  float    v8f;

#define BATCH   128
#define EPS     1e-7f
#define N_CAPS1 1152
#define D_CAPS1 8
#define N_CAPS2 10
#define D_CAPS2 16

// conv2 implicit GEMM dims
#define M2 (BATCH * 36)      // 4608
#define N2 256
#define K2 (9 * 9 * 256)     // 20736
#define KC (K2 / 32)         // 648 chunks of 32

// ---------------------------------------------------------------------------
// conv1: 28x28x1 -> 20x20x256 (9x9, stride1) + bias + relu, output f16 NHWC
// ---------------------------------------------------------------------------
__global__ void k_conv1(const float* __restrict__ img, const float* __restrict__ w,
                        const float* __restrict__ bias, _Float16* __restrict__ out) {
    int tid = blockIdx.x * blockDim.x + threadIdx.x;
    if (tid >= BATCH * 20 * 20 * 256) return;
    int co = tid & 255;
    int t  = tid >> 8;
    int ox = t % 20; t /= 20;
    int oy = t % 20; t /= 20;
    int b  = t;
    const float* ip = img + b * 28 * 28;   // single input channel
    float acc = bias[co];
    #pragma unroll
    for (int ky = 0; ky < 9; ++ky)
        for (int kx = 0; kx < 9; ++kx)
            acc += ip[(oy + ky) * 28 + (ox + kx)] * w[(ky * 9 + kx) * 256 + co];
    out[tid] = (_Float16)fmaxf(acc, 0.f);
}

// ---------------------------------------------------------------------------
// pack conv2 weights (HWIO == row-major [K2][N2]) into per-lane WMMA B layout:
// element index = ((kc*256 + n)*2 + half)*16 + e, where K = kc*32 + (e>>3)*16 + half*8 + (e&7)
// For fixed kc, a 64-column group is a contiguous 4KB block -> LDS staging.
// ---------------------------------------------------------------------------
__global__ void k_packw(const float* __restrict__ w2, _Float16* __restrict__ bpk) {
    int tid = blockIdx.x * blockDim.x + threadIdx.x;
    if (tid >= KC * 256 * 32) return;
    int e  = tid & 15;
    int h  = (tid >> 4) & 1;
    int n  = (tid >> 5) & 255;
    int kc = tid >> 13;
    int K  = kc * 32 + (e >> 3) * 16 + h * 8 + (e & 7);
    bpk[tid] = (_Float16)w2[K * 256 + n];
}

// ---------------------------------------------------------------------------
// conv2 as implicit GEMM via v_wmma_f32_16x16x32_f16.
// Block (4 waves) computes a 64x64 tile: wave w -> rows [w*16, w*16+16),
// 4 accumulators over the 64 N-columns. B tile staged through LDS (4KB/chunk).
// ---------------------------------------------------------------------------
__global__ void k_conv2_wmma(const _Float16* __restrict__ A, const _Float16* __restrict__ Bp,
                             const float* __restrict__ bias, float* __restrict__ out) {
    __shared__ __attribute__((aligned(32))) _Float16 smem[32 * 64];  // 4KB B tile, packed lane layout
    int t    = threadIdx.x;              // 0..127
    int lane = t & 31;
    int wave = t >> 5;
    int mB = blockIdx.x >> 2;            // 0..71  (M2/64)
    int nB = blockIdx.x & 3;             // 0..3   (N2/64)
    int h   = lane >> 4;                 // half: 0/1
    int l16 = lane & 15;

    // A row for this lane (ISA A-layout: lanes 0-15 and 16-31 both carry M=l16)
    int m  = mB * 64 + wave * 16 + l16;
    int b  = m / 36;
    int r  = m % 36;
    int oy = r / 6, ox = r % 6;

    int seg = t * 16;                    // per-thread 16-half (32B) staging segment
    v8f acc[4] = {{}, {}, {}, {}};

    for (int kc = 0; kc < KC; ++kc) {
        int kk = kc * 32;
        int p  = kk >> 8;                // filter tap 0..80 (each tap = 256 channels)
        int ky = p / 9, kx = p % 9;
        int c0 = kk & 255;
        // A fragment: K(e) = 16*(e>>3) + 8*h + (e&7) -> two contiguous 16B loads
        int aidx = ((b * 20 + (oy * 2 + ky)) * 20 + (ox * 2 + kx)) * 256 + c0 + h * 8;

        // cooperative B staging: contiguous 4KB = 2048 halves for (kc, 64 cols)
        const _Float16* bsrc = Bp + ((size_t)(kc * 256 + nB * 64) * 2) * 16 + seg;
        __syncthreads();
        *(v8h*)(smem + seg)     = *(const v8h*)bsrc;
        *(v8h*)(smem + seg + 8) = *(const v8h*)(bsrc + 8);
        if (kc + 1 < KC)
            __builtin_prefetch(bsrc + 256 * 32, 0, 0);   // next chunk's 4KB block
        __syncthreads();

        v8h alo = *(const v8h*)(A + aidx);
        v8h ahi = *(const v8h*)(A + aidx + 16);
        v16h av = __builtin_shufflevector(alo, ahi,
                    0, 1, 2, 3, 4, 5, 6, 7, 8, 9, 10, 11, 12, 13, 14, 15);
        #pragma unroll
        for (int s = 0; s < 4; ++s) {
            // fragment for n-subtile s: 32B contiguous in LDS
            v16h bv = *(const v16h*)(smem + ((s * 16 + l16) * 2 + h) * 16);
            acc[s] = __builtin_amdgcn_wmma_f32_16x16x32_f16(
                         false, av, false, bv, (short)0, acc[s], false, false);
        }
    }
    // C/D layout: VGPR rr holds M = rr + 8*h, N = l16. Add bias + relu.
    #pragma unroll
    for (int s = 0; s < 4; ++s) {
        int n = nB * 64 + s * 16 + l16;
        float bb = bias[n];
        #pragma unroll
        for (int rr = 0; rr < 8; ++rr) {
            int mm = mB * 64 + wave * 16 + rr + 8 * h;
            out[(size_t)mm * N2 + n] = fmaxf(acc[s][rr] + bb, 0.f);
        }
    }
}

// ---------------------------------------------------------------------------
// primary capsule squash: [B,1152,8] over last axis
// ---------------------------------------------------------------------------
__global__ void k_squash_pc(const float* __restrict__ x, float* __restrict__ pc) {
    int tid = blockIdx.x * blockDim.x + threadIdx.x;
    if (tid >= BATCH * N_CAPS1) return;
    const float* s = x + (size_t)tid * 8;
    float sq = 0.f;
    #pragma unroll
    for (int d = 0; d < 8; ++d) sq += s[d] * s[d];
    float scale = (sq / (1.f + sq)) * rsqrtf(sq + EPS);
    #pragma unroll
    for (int d = 0; d < 8; ++d) pc[(size_t)tid * 8 + d] = s[d] * scale;
}

// ---------------------------------------------------------------------------
// routing round 1: s1[b,j,d] = 0.1 * sum_i u_hat   (softmax of zeros = 1/10)
// ---------------------------------------------------------------------------
__global__ void k_s1(const float* __restrict__ W, const float* __restrict__ pc,
                     float* __restrict__ s1) {
    int tid = blockIdx.x * blockDim.x + threadIdx.x;
    if (tid >= BATCH * N_CAPS2 * D_CAPS2) return;
    int d = tid & 15;
    int j = (tid >> 4) % 10;
    int b = tid / 160;
    float acc = 0.f;
    for (int i = 0; i < N_CAPS1; ++i) {
        const float* wr = W + (size_t)((i * 10 + j) * 16 + d) * 8;
        const float* pr = pc + (size_t)(b * N_CAPS1 + i) * 8;
        float u = 0.f;
        #pragma unroll
        for (int k = 0; k < 8; ++k) u += wr[k] * pr[k];
        acc += u;
    }
    s1[tid] = acc * 0.1f;
}

// squash over 16-element vectors (axis = -2 of [B,1,10,16,1])
__global__ void k_squash16(const float* __restrict__ s, float* __restrict__ v) {
    int tid = blockIdx.x * blockDim.x + threadIdx.x;
    if (tid >= BATCH * N_CAPS2) return;
    const float* p = s + (size_t)tid * 16;
    float sq = 0.f;
    #pragma unroll
    for (int d = 0; d < 16; ++d) sq += p[d] * p[d];
    float scale = (sq / (1.f + sq)) * rsqrtf(sq + EPS);
    #pragma unroll
    for (int d = 0; d < 16; ++d) v[(size_t)tid * 16 + d] = p[d] * scale;
}

// ---------------------------------------------------------------------------
// routing round 2: raw[b,i,j] = <u_hat, v1>, c2 = softmax_j(raw)
// ---------------------------------------------------------------------------
__global__ void k_route(const float* __restrict__ W, const float* __restrict__ pc,
                        const float* __restrict__ v1, float* __restrict__ c2) {
    int tid = blockIdx.x * blockDim.x + threadIdx.x;
    if (tid >= BATCH * N_CAPS1) return;
    int b = tid / N_CAPS1;
    int i = tid % N_CAPS1;
    const float* pr = pc + (size_t)tid * 8;
    float raw[N_CAPS2];
    float mx = -1e30f;
    for (int j = 0; j < N_CAPS2; ++j) {
        float uv = 0.f;
        for (int d = 0; d < 16; ++d) {
            const float* wr = W + (size_t)((i * 10 + j) * 16 + d) * 8;
            float u = 0.f;
            #pragma unroll
            for (int k = 0; k < 8; ++k) u += wr[k] * pr[k];
            uv += u * v1[(size_t)(b * 10 + j) * 16 + d];
        }
        raw[j] = uv;
        mx = fmaxf(mx, uv);
    }
    float sum = 0.f;
    for (int j = 0; j < N_CAPS2; ++j) { raw[j] = expf(raw[j] - mx); sum += raw[j]; }
    float inv = 1.f / sum;
    for (int j = 0; j < N_CAPS2; ++j) c2[(size_t)tid * 10 + j] = raw[j] * inv;
}

// s2[b,j,d] = sum_i c2[b,i,j] * u_hat[b,i,j,d]   (u_hat recomputed)
__global__ void k_s2(const float* __restrict__ W, const float* __restrict__ pc,
                     const float* __restrict__ c2, float* __restrict__ s2) {
    int tid = blockIdx.x * blockDim.x + threadIdx.x;
    if (tid >= BATCH * N_CAPS2 * D_CAPS2) return;
    int d = tid & 15;
    int j = (tid >> 4) % 10;
    int b = tid / 160;
    float acc = 0.f;
    for (int i = 0; i < N_CAPS1; ++i) {
        const float* wr = W + (size_t)((i * 10 + j) * 16 + d) * 8;
        const float* pr = pc + (size_t)(b * N_CAPS1 + i) * 8;
        float u = 0.f;
        #pragma unroll
        for (int k = 0; k < 8; ++k) u += wr[k] * pr[k];
        acc += c2[(size_t)(b * N_CAPS1 + i) * 10 + j] * u;
    }
    s2[tid] = acc;
}

// v2 = squash(s2); norm_v2 -> d_out; masked v2 -> decoder input h0
__global__ void k_v2(const float* __restrict__ s2, const int* __restrict__ target,
                     float* __restrict__ norm_out, float* __restrict__ h0) {
    int tid = blockIdx.x * blockDim.x + threadIdx.x;
    if (tid >= BATCH * N_CAPS2) return;
    int b = tid / 10, j = tid % 10;
    const float* p = s2 + (size_t)tid * 16;
    float sq = 0.f;
    #pragma unroll
    for (int d = 0; d < 16; ++d) sq += p[d] * p[d];
    float scale = (sq / (1.f + sq)) * rsqrtf(sq + EPS);
    float v2sq = 0.f;
    float msk = (target[b] == j) ? 1.f : 0.f;
    #pragma unroll
    for (int d = 0; d < 16; ++d) {
        float v = p[d] * scale;
        v2sq += v * v;
        h0[(size_t)b * 160 + j * 16 + d] = v * msk;
    }
    norm_out[b * 10 + j] = sqrtf(v2sq + EPS);
}

__global__ void k_argmax(const float* __restrict__ norm, float* __restrict__ pred) {
    int b = blockIdx.x * blockDim.x + threadIdx.x;
    if (b >= BATCH) return;
    float best = norm[b * 10];
    int bi = 0;
    for (int j = 1; j < 10; ++j) {
        float v = norm[b * 10 + j];
        if (v > best) { best = v; bi = j; }
    }
    pred[b] = (float)bi;
}

// generic dense layer: act 0=relu, 1=sigmoid
__global__ void k_dense(const float* __restrict__ x, const float* __restrict__ w,
                        const float* __restrict__ bias, float* __restrict__ y,
                        int In, int Out, int act) {
    int tid = blockIdx.x * blockDim.x + threadIdx.x;
    if (tid >= BATCH * Out) return;
    int o = tid % Out, b = tid / Out;
    float acc = bias[o];
    for (int k = 0; k < In; ++k) acc += x[(size_t)b * In + k] * w[(size_t)k * Out + o];
    y[tid] = (act == 0) ? fmaxf(acc, 0.f) : (1.f / (1.f + expf(-acc)));
}

// ---------------------------------------------------------------------------
extern "C" void kernel_launch(void* const* d_in, const int* in_sizes, int n_in,
                              void* d_out, int out_size, void* d_ws, size_t ws_size,
                              hipStream_t stream) {
    const float* image   = (const float*)d_in[0];
    const int*   target  = (const int*)  d_in[1];
    const float* conv1_w = (const float*)d_in[2];
    const float* conv1_b = (const float*)d_in[3];
    const float* conv2_w = (const float*)d_in[4];
    const float* conv2_b = (const float*)d_in[5];
    const float* W       = (const float*)d_in[6];
    const float* d1_w    = (const float*)d_in[7];
    const float* d1_b    = (const float*)d_in[8];
    const float* d2_w    = (const float*)d_in[9];
    const float* d2_b    = (const float*)d_in[10];
    const float* do_w    = (const float*)d_in[11];
    const float* do_b    = (const float*)d_in[12];

    char* ws = (char*)d_ws;
    size_t oA  = 0;                                 // conv1 out f16: 128*20*20*256
    size_t oBp = oA  + (size_t)BATCH * 20 * 20 * 256 * 2;   // packed w2 f16
    size_t oC2 = oBp + (size_t)KC * 256 * 32 * 2;           // conv2 out f32
    size_t oPC = oC2 + (size_t)M2 * N2 * 4;                 // pc f32
    size_t oS1 = oPC + (size_t)BATCH * N_CAPS1 * 8 * 4;
    size_t oV1 = oS1 + (size_t)BATCH * 160 * 4;
    size_t oCC = oV1 + (size_t)BATCH * 160 * 4;             // c2 [B,1152,10]
    size_t oS2 = oCC + (size_t)BATCH * N_CAPS1 * 10 * 4;
    size_t oH0 = oS2 + (size_t)BATCH * 160 * 4;
    size_t oH1 = oH0 + (size_t)BATCH * 160 * 4;
    size_t oH2 = oH1 + (size_t)BATCH * 512 * 4;

    _Float16* a16 = (_Float16*)(ws + oA);
    _Float16* bpk = (_Float16*)(ws + oBp);
    float* c2out  = (float*)(ws + oC2);
    float* pc     = (float*)(ws + oPC);
    float* s1     = (float*)(ws + oS1);
    float* v1     = (float*)(ws + oV1);
    float* cc     = (float*)(ws + oCC);
    float* s2     = (float*)(ws + oS2);
    float* h0     = (float*)(ws + oH0);
    float* h1     = (float*)(ws + oH1);
    float* h2     = (float*)(ws + oH2);

    float* out_norm = (float*)d_out;            // 1280
    float* out_pred = out_norm + 1280;          // 128
    float* out_dec  = out_pred + 128;           // 100352

    k_conv1<<<(BATCH * 20 * 20 * 256 + 255) / 256, 256, 0, stream>>>(image, conv1_w, conv1_b, a16);
    k_packw<<<(KC * 256 * 32 + 255) / 256, 256, 0, stream>>>(conv2_w, bpk);
    k_conv2_wmma<<<(M2 / 64) * (N2 / 64), 128, 0, stream>>>(a16, bpk, conv2_b, c2out);
    k_squash_pc<<<(BATCH * N_CAPS1 + 255) / 256, 256, 0, stream>>>(c2out, pc);
    k_s1<<<(BATCH * 160 + 255) / 256, 256, 0, stream>>>(W, pc, s1);
    k_squash16<<<(BATCH * 10 + 255) / 256, 256, 0, stream>>>(s1, v1);
    k_route<<<(BATCH * N_CAPS1 + 255) / 256, 256, 0, stream>>>(W, pc, v1, cc);
    k_s2<<<(BATCH * 160 + 255) / 256, 256, 0, stream>>>(W, pc, cc, s2);
    k_v2<<<(BATCH * 10 + 255) / 256, 256, 0, stream>>>(s2, target, out_norm, h0);
    k_argmax<<<1, 128, 0, stream>>>(out_norm, out_pred);
    k_dense<<<(BATCH * 512 + 255) / 256, 256, 0, stream>>>(h0, d1_w, d1_b, h1, 160, 512, 0);
    k_dense<<<(BATCH * 1024 + 255) / 256, 256, 0, stream>>>(h1, d2_w, d2_b, h2, 512, 1024, 0);
    k_dense<<<(BATCH * 784 + 255) / 256, 256, 0, stream>>>(h2, do_w, do_b, out_dec, 1024, 784, 1);
}
